// AttentionRope_36481452212572
// MI455X (gfx1250) — matrix operand order repaired and checked by reading, not compiled, not run on hardware
//
#include <hip/hip_runtime.h>
#include <hip/hip_bf16.h>

#if __has_builtin(__builtin_amdgcn_global_load_async_to_lds_b128) && \
    __has_builtin(__builtin_amdgcn_s_wait_asynccnt)
#define USE_ASYNC_LDS 1
#else
#define USE_ASYNC_LDS 0
#endif

namespace {

constexpr int   kBa   = 8;
constexpr int   kN    = 1025;
constexpr int   kC    = 1024;
constexpr int   kH    = 16;
constexpr int   kHD   = 64;
constexpr int   kNPad = 1056;   // 33 * 32, padded token count for K/V and Q rows
constexpr int   kRowT = 17;     // ceil(1025 / 64)
constexpr int   kQT   = 65;     // ceil(1025 / 16)
constexpr float kScale = 0.125f; // 64^-0.5
constexpr float kEps   = 1e-6f;

typedef __attribute__((ext_vector_type(16))) __bf16 v16bf;
typedef __attribute__((ext_vector_type(8)))  float  v8f;
typedef int vi4 __attribute__((vector_size(16)));   // matches async builtin param

union Frag {
  v16bf v;
  uint4 q[2];
};

__device__ inline v8f zero8() {
  v8f z;
#pragma unroll
  for (int i = 0; i < 8; ++i) z[i] = 0.f;
  return z;
}

// ---------------------------------------------------------------------------
// Kernel 1: fused QKV GEMM + bias + QK-LayerNorm + RoPE + scale, bf16 outputs.
// Grid: (17 row-tiles, 48 col-tiles(mat*16+head), 8 batch). Block: 128 (4 waves).
// Double-buffered LDS, one barrier per k-step.
// ---------------------------------------------------------------------------
__global__ __launch_bounds__(128) void qkv_ln_rope_kernel(
    const float* __restrict__ x, const float* __restrict__ rope,
    const float* __restrict__ qkv_w, const float* __restrict__ qkv_b,
    const float* __restrict__ qn_w, const float* __restrict__ qn_b,
    const float* __restrict__ kn_w, const float* __restrict__ kn_b,
    __bf16* __restrict__ qb, __bf16* __restrict__ kb, __bf16* __restrict__ vt)
{
  const int b    = blockIdx.z;
  const int n0   = blockIdx.x * 64;
  const int cb   = blockIdx.y;          // 0..47 -> (mat, head)
  const int mat  = cb / kH;             // 0=q, 1=k, 2=v
  const int head = cb % kH;
  const int c0   = cb * 64;             // column base in [0, 3C)

  const int t    = threadIdx.x;
  const int lane = t & 31;
  const int wv   = t >> 5;
  const int hi   = (lane >> 4) & 1;
  const int ln   = lane & 15;

  __shared__ __bf16 Al[2][64][40];      // A tiles (rows x k), padded stride
  __shared__ __bf16 Bl[2][64][40];      // B tiles (cols x k), padded stride
  __shared__ float  Sd[64][65];         // f32 result tile for epilogue

  v8f acc[4];
#pragma unroll
  for (int s = 0; s < 4; ++s) acc[s] = zero8();

  // Batched load: all global loads issue before any conversion consumes them.
  auto loadTiles = [&](int k0, float4 (&ax)[4], float4 (&bw)[4]) {
#pragma unroll
    for (int i = 0; i < 4; ++i) {
      int e  = t + i * 128;             // 0..511 chunks of 4 floats
      int r  = e >> 3;
      int cg = (e & 7) * 4;
      int n  = n0 + r;
      ax[i] = make_float4(0.f, 0.f, 0.f, 0.f);
      if (n < kN)
        ax[i] = *(const float4*)(x + ((size_t)b * kN + n) * kC + k0 + cg);
      bw[i] = *(const float4*)(qkv_w + (size_t)(c0 + r) * kC + k0 + cg);
    }
  };
  auto storeTiles = [&](const float4 (&ax)[4], const float4 (&bw)[4], int buf) {
#pragma unroll
    for (int i = 0; i < 4; ++i) {
      int e  = t + i * 128;
      int r  = e >> 3;
      int cg = (e & 7) * 4;
      Al[buf][r][cg + 0] = (__bf16)ax[i].x;
      Al[buf][r][cg + 1] = (__bf16)ax[i].y;
      Al[buf][r][cg + 2] = (__bf16)ax[i].z;
      Al[buf][r][cg + 3] = (__bf16)ax[i].w;
      Bl[buf][r][cg + 0] = (__bf16)bw[i].x;
      Bl[buf][r][cg + 1] = (__bf16)bw[i].y;
      Bl[buf][r][cg + 2] = (__bf16)bw[i].z;
      Bl[buf][r][cg + 3] = (__bf16)bw[i].w;
    }
  };

  float4 ax[4], bw[4];
  loadTiles(0, ax, bw);
  int cur = 0;
  for (int k0 = 0; k0 < kC; k0 += 32) {
    storeTiles(ax, bw, cur);
    if (k0 + 32 < kC) {
      loadTiles(k0 + 32, ax, bw);
      if (k0 + 64 < kC)  // stream-prefetch weights two k-steps ahead
        __builtin_prefetch(qkv_w + (size_t)(c0 + (t >> 1)) * kC + k0 + 64 +
                               ((t & 1) << 4), 0, 1);
    }
    __syncthreads();

    Frag a;
    {
      int row = wv * 16 + ln;
      int o   = hi ? 16 : 0;            // A layout: 16 contiguous k per lane
      a.q[0] = *(const uint4*)&Al[cur][row][o];
      a.q[1] = *(const uint4*)&Al[cur][row][o + 8];
    }
#pragma unroll
    for (int s = 0; s < 4; ++s) {
      Frag bf;
      int col = s * 16 + ln;
      int o   = hi ? 16 : 0;            // B layout: 16 contiguous k per lane
      bf.q[0] = *(const uint4*)&Bl[cur][col][o];
      bf.q[1] = *(const uint4*)&Bl[cur][col][o + 8];
      acc[s] = __builtin_amdgcn_wmma_f32_16x16x32_bf16(
          false, a.v, false, bf.v, (short)0, acc[s], false, false);
    }
    cur ^= 1;
  }

  // Dump tile (+bias) to LDS for row-wise epilogue
#pragma unroll
  for (int s = 0; s < 4; ++s) {
    int col    = s * 16 + ln;
    float bias = qkv_b[c0 + col];
#pragma unroll
    for (int r = 0; r < 8; ++r) {
      int row = wv * 16 + r + hi * 8;
      Sd[row][col] = acc[s][r] + bias;
    }
  }
  __syncthreads();

  if (mat == 2) {
    // V: transposed store vt[(bh*64 + d)*NPad + n]
    int d  = t >> 1;
    int j0 = (t & 1) * 32;
    __bf16* dst = vt + ((size_t)(b * kH + head) * 64 + d) * kNPad;
    for (int j = 0; j < 32; ++j) {
      int n = n0 + j0 + j;
      if (n < kN) dst[n] = (__bf16)Sd[j0 + j][d];
    }
  } else if (t < 64) {
    int n = n0 + t;
    if (n < kN) {
      float buf[64];
      float mu = 0.f;
#pragma unroll
      for (int d = 0; d < 64; ++d) { buf[d] = Sd[t][d]; mu += buf[d]; }
      mu *= (1.f / 64.f);
      float var = 0.f;
#pragma unroll
      for (int d = 0; d < 64; ++d) { float z = buf[d] - mu; var += z * z; }
      var *= (1.f / 64.f);
      float rs = rsqrtf(var + kEps);
      const float* w  = (mat == 0) ? qn_w : kn_w;
      const float* bs = (mat == 0) ? qn_b : kn_b;
#pragma unroll
      for (int d = 0; d < 64; ++d) buf[d] = (buf[d] - mu) * rs * w[d] + bs[d];
      if (n >= 1) {  // RoPE (cls token passes through)
        const float* rp = rope + (size_t)(n - 1) * 128;  // [sin(64) | cos(64)]
#pragma unroll
        for (int d = 0; d < 64; d += 2) {
          float s0 = rp[d], s1 = rp[d + 1];
          float cA = rp[64 + d], cB = rp[64 + d + 1];
          float x0 = buf[d], x1 = buf[d + 1];
          buf[d]     = x0 * cA - x1 * s0;
          buf[d + 1] = x1 * cB + x0 * s1;
        }
      }
      float sc = (mat == 0) ? kScale : 1.f;
      __bf16* dst = ((mat == 0) ? qb : kb) +
                    ((size_t)(b * kH + head) * kNPad + n) * 64;
#pragma unroll
      for (int d = 0; d < 64; ++d) dst[d] = (__bf16)(buf[d] * sc);
    }
  }
}

// ---------------------------------------------------------------------------
// Kernel 2: flash attention, one wave per (b, h, 16-query tile).
// K and V fragments are issued up front so V loads overlap the softmax VALU.
// ---------------------------------------------------------------------------
__global__ __launch_bounds__(32) void attn_kernel(
    const __bf16* __restrict__ qb, const __bf16* __restrict__ kb,
    const __bf16* __restrict__ vt, __bf16* __restrict__ ob)
{
  const int qt   = blockIdx.x;          // 0..64
  const int h    = blockIdx.y;
  const int b    = blockIdx.z;
  const int lane = threadIdx.x;
  const int hi   = (lane >> 4) & 1;
  const int ln   = lane & 15;
  const int q0   = qt * 16;

  const __bf16* qs = qb + (size_t)(b * kH + h) * kNPad * 64;
  const __bf16* ks = kb + (size_t)(b * kH + h) * kNPad * 64;
  const __bf16* vs = vt + (size_t)(b * kH + h) * 64 * kNPad;

  __shared__ __bf16 Pl[16][40];         // P tile (16 q x 32 k), padded

  // Q fragments (two K=32 steps over head_dim)
  Frag qf[2];
  {
    const __bf16* base = qs + (size_t)(q0 + ln) * 64;
    int o = hi ? 16 : 0;
#pragma unroll
    for (int s = 0; s < 2; ++s) {
      qf[s].q[0] = *(const uint4*)(base + s * 32 + o);
      qf[s].q[1] = *(const uint4*)(base + s * 32 + o + 8);
    }
  }

  float mrow[8], lrow[8];
  v8f oacc[4];
#pragma unroll
  for (int r = 0; r < 8; ++r) { mrow[r] = -1e30f; lrow[r] = 0.f; }
#pragma unroll
  for (int s = 0; s < 4; ++s) oacc[s] = zero8();

  for (int kb0 = 0; kb0 < kNPad; kb0 += 32) {
    // Issue every global load of this block up front.
    Frag kf[2][2];
    Frag vf[4];
#pragma unroll
    for (int sub = 0; sub < 2; ++sub) {
      const __bf16* kbase = ks + (size_t)(kb0 + sub * 16 + ln) * 64;
#pragma unroll
      for (int kst = 0; kst < 2; ++kst) {
        int dstart = kst * 32 + (hi ? 16 : 0);
        kf[sub][kst].q[0] = *(const uint4*)(kbase + dstart);
        kf[sub][kst].q[1] = *(const uint4*)(kbase + dstart + 8);
      }
    }
#pragma unroll
    for (int dt = 0; dt < 4; ++dt) {
      const __bf16* vb = vs + (size_t)(dt * 16 + ln) * kNPad + kb0 + (hi ? 16 : 0);
      vf[dt].q[0] = *(const uint4*)(vb);
      vf[dt].q[1] = *(const uint4*)(vb + 8);
    }

    // --- S tiles ---
    v8f st[2];
#pragma unroll
    for (int sub = 0; sub < 2; ++sub) {
      v8f s = zero8();
#pragma unroll
      for (int kst = 0; kst < 2; ++kst)
        s = __builtin_amdgcn_wmma_f32_16x16x32_bf16(
            false, qf[kst].v, false, kf[sub][kst].v, (short)0, s, false, false);
      st[sub] = s;
    }
    // Mask keys >= N (lane-uniform: lane owns one key column per subtile)
#pragma unroll
    for (int sub = 0; sub < 2; ++sub) {
      int col = kb0 + sub * 16 + ln;
      if (col >= kN) {
#pragma unroll
        for (int r = 0; r < 8; ++r) st[sub][r] = -1e30f;
      }
    }
    // --- Online softmax: per-row stats via intra-half shuffles ---
#pragma unroll
    for (int r = 0; r < 8; ++r) {
      float v = fmaxf(st[0][r], st[1][r]);
#pragma unroll
      for (int m = 1; m < 16; m <<= 1) v = fmaxf(v, __shfl_xor(v, m, 32));
      float mn = fmaxf(mrow[r], v);
      float p0 = __expf(st[0][r] - mn);
      float p1 = __expf(st[1][r] - mn);
      st[0][r] = p0; st[1][r] = p1;
      float ps = p0 + p1;
#pragma unroll
      for (int m = 1; m < 16; m <<= 1) ps += __shfl_xor(ps, m, 32);
      float sc = __expf(mrow[r] - mn);
      lrow[r]  = lrow[r] * sc + ps;
      mrow[r]  = mn;
#pragma unroll
      for (int s = 0; s < 4; ++s) oacc[s][r] *= sc;
    }
    // --- P through LDS: C-layout -> A-layout ---
#pragma unroll
    for (int sub = 0; sub < 2; ++sub)
#pragma unroll
      for (int r = 0; r < 8; ++r)
        Pl[r + hi * 8][sub * 16 + ln] = (__bf16)st[sub][r];
    __syncthreads();

    Frag pf;
    {
      int o = hi ? 16 : 0;
      pf.q[0] = *(const uint4*)&Pl[ln][o];
      pf.q[1] = *(const uint4*)&Pl[ln][o + 8];
    }
#pragma unroll
    for (int dt = 0; dt < 4; ++dt)
      oacc[dt] = __builtin_amdgcn_wmma_f32_16x16x32_bf16(
          false, pf.v, false, vf[dt].v, (short)0, oacc[dt], false, false);
    __syncthreads();
  }

  // Normalize and store valid rows
#pragma unroll
  for (int dt = 0; dt < 4; ++dt)
#pragma unroll
    for (int r = 0; r < 8; ++r) {
      int n = q0 + r + hi * 8;
      if (n < kN) {
        float val = oacc[dt][r] / lrow[r];
        ob[((size_t)b * kN + n) * kC + h * 64 + dt * 16 + ln] = (__bf16)val;
      }
    }
}

// ---------------------------------------------------------------------------
// Kernel 3: output projection GEMM: out = ob @ proj_w^T + proj_b (f32 out).
// A-tile (already bf16) staged via GLOBAL_LOAD_ASYNC_TO_LDS when available;
// B-tile converted f32->bf16 through registers. Double-buffered LDS.
// ---------------------------------------------------------------------------
__global__ __launch_bounds__(128) void proj_kernel(
    const __bf16* __restrict__ ob, const float* __restrict__ proj_w,
    const float* __restrict__ proj_b, float* __restrict__ out)
{
  const int b  = blockIdx.z;
  const int n0 = blockIdx.x * 64;
  const int c0 = blockIdx.y * 64;

  const int t    = threadIdx.x;
  const int lane = t & 31;
  const int wv   = t >> 5;
  const int hi   = (lane >> 4) & 1;
  const int ln   = lane & 15;

  __shared__ __bf16 Al[2][64][40];
  __shared__ __bf16 Bl[2][64][40];

  v8f acc[4];
#pragma unroll
  for (int s = 0; s < 4; ++s) acc[s] = zero8();

#if USE_ASYNC_LDS
  // Async DMA of the bf16 A-tile straight into LDS (rows >= N clamped; those
  // tile rows only feed output rows that are never stored).
  auto asyncA = [&](int k0, int buf) {
#pragma unroll
    for (int i = 0; i < 2; ++i) {
      int e  = t + i * 128;             // 0..255 chunks of 8 bf16 (16B)
      int r  = e >> 2;
      int cg = (e & 3) * 8;
      int n  = n0 + r; if (n >= kN) n = kN - 1;
      const __bf16* g = ob + ((size_t)b * kN + n) * kC + k0 + cg;
      __builtin_amdgcn_global_load_async_to_lds_b128(
          (__attribute__((address_space(1))) vi4*)(vi4*)g,
          (__attribute__((address_space(3))) vi4*)&Al[buf][r][cg], 0, 0);
    }
  };
#else
  auto loadA = [&](int k0, uint4 (&aw)[2]) {
#pragma unroll
    for (int i = 0; i < 2; ++i) {
      int e  = t + i * 128;
      int r  = e >> 2;
      int cg = (e & 3) * 8;
      int n  = n0 + r;
      aw[i] = make_uint4(0u, 0u, 0u, 0u);
      if (n < kN)
        aw[i] = *(const uint4*)(ob + ((size_t)b * kN + n) * kC + k0 + cg);
    }
  };
  auto storeA = [&](const uint4 (&aw)[2], int buf) {
#pragma unroll
    for (int i = 0; i < 2; ++i) {
      int e  = t + i * 128;
      int r  = e >> 2;
      int cg = (e & 3) * 8;
      *(uint4*)&Al[buf][r][cg] = aw[i];
    }
  };
#endif
  auto loadB = [&](int k0, float4 (&bwr)[4]) {
#pragma unroll
    for (int i = 0; i < 4; ++i) {
      int e   = t + i * 128;
      int col = e >> 3;
      int kk  = (e & 7) * 4;
      bwr[i] = *(const float4*)(proj_w + (size_t)(c0 + col) * kC + k0 + kk);
    }
  };
  auto storeB = [&](const float4 (&bwr)[4], int buf) {
#pragma unroll
    for (int i = 0; i < 4; ++i) {
      int e   = t + i * 128;
      int col = e >> 3;
      int kk  = (e & 7) * 4;
      Bl[buf][col][kk + 0] = (__bf16)bwr[i].x;
      Bl[buf][col][kk + 1] = (__bf16)bwr[i].y;
      Bl[buf][col][kk + 2] = (__bf16)bwr[i].z;
      Bl[buf][col][kk + 3] = (__bf16)bwr[i].w;
    }
  };

  float4 bwr[4];
#if USE_ASYNC_LDS
  asyncA(0, 0);
#else
  uint4 aw[2];
  loadA(0, aw);
#endif
  loadB(0, bwr);

  int cur = 0;
  for (int k0 = 0; k0 < kC; k0 += 32) {
    const bool more = (k0 + 32 < kC);
    storeB(bwr, cur);
#if !USE_ASYNC_LDS
    storeA(aw, cur);
    if (more) loadA(k0 + 32, aw);
#endif
    if (more) {
      loadB(k0 + 32, bwr);
      if (k0 + 64 < kC)
        __builtin_prefetch(proj_w + (size_t)(c0 + (t >> 1)) * kC + k0 + 64 +
                               ((t & 1) << 4), 0, 1);
    }
#if USE_ASYNC_LDS
    __builtin_amdgcn_s_wait_asynccnt(0);  // my Al[cur] DMA writes have landed
#endif
    __syncthreads();
#if USE_ASYNC_LDS
    if (more) asyncA(k0 + 32, cur ^ 1);   // safe: cur^1 readers finished pre-barrier
#endif

    Frag a;
    {
      int row = wv * 16 + ln;
      int o   = hi ? 16 : 0;
      a.q[0] = *(const uint4*)&Al[cur][row][o];
      a.q[1] = *(const uint4*)&Al[cur][row][o + 8];
    }
#pragma unroll
    for (int s = 0; s < 4; ++s) {
      Frag bf;
      int col = s * 16 + ln;
      int o   = hi ? 16 : 0;
      bf.q[0] = *(const uint4*)&Bl[cur][col][o];
      bf.q[1] = *(const uint4*)&Bl[cur][col][o + 8];
      acc[s] = __builtin_amdgcn_wmma_f32_16x16x32_bf16(
          false, a.v, false, bf.v, (short)0, acc[s], false, false);
    }
    cur ^= 1;
  }

#pragma unroll
  for (int s = 0; s < 4; ++s) {
    int col    = c0 + s * 16 + ln;
    float bias = proj_b[col];
#pragma unroll
    for (int r = 0; r < 8; ++r) {
      int n = n0 + wv * 16 + r + hi * 8;
      if (n < kN)
        out[((size_t)b * kN + n) * kC + col] = acc[s][r] + bias;
    }
  }
}

} // namespace

extern "C" void kernel_launch(void* const* d_in, const int* in_sizes, int n_in,
                              void* d_out, int out_size, void* d_ws, size_t ws_size,
                              hipStream_t stream) {
  (void)in_sizes; (void)n_in; (void)out_size; (void)ws_size;
  const float* x      = (const float*)d_in[0];
  const float* rope   = (const float*)d_in[1];
  const float* qkv_w  = (const float*)d_in[2];
  const float* qkv_b  = (const float*)d_in[3];
  const float* qn_w   = (const float*)d_in[4];
  const float* qn_b   = (const float*)d_in[5];
  const float* kn_w   = (const float*)d_in[6];
  const float* kn_b   = (const float*)d_in[7];
  const float* proj_w = (const float*)d_in[8];
  const float* proj_b = (const float*)d_in[9];

  char* ws = (char*)d_ws;
  const size_t szQKV = (size_t)kBa * kH * kNPad * kHD * sizeof(__bf16); // 17.3 MB
  __bf16* qb = (__bf16*)(ws);
  __bf16* kb = (__bf16*)(ws + szQKV);
  __bf16* vt = (__bf16*)(ws + 2 * szQKV);
  __bf16* ob = (__bf16*)(ws + 3 * szQKV);

  // Zero q/k/v (covers the NPad tail rows read by the attention kernel).
  (void)hipMemsetAsync(ws, 0, 3 * szQKV, stream);

  dim3 g1(kRowT, 48, kBa);
  qkv_ln_rope_kernel<<<g1, dim3(128), 0, stream>>>(
      x, rope, qkv_w, qkv_b, qn_w, qn_b, kn_w, kn_b, qb, kb, vt);

  dim3 g2(kQT, kH, kBa);
  attn_kernel<<<g2, dim3(32), 0, stream>>>(qb, kb, vt, ob);

  dim3 g3(kRowT, kC / 64, kBa);
  proj_kernel<<<g3, dim3(128), 0, stream>>>(ob, proj_w, proj_b, (float*)d_out);
}
